// Hard_sim_sample_generator_59055800320097
// MI455X (gfx1250) — matrix operand order repaired and checked by reading, not compiled, not run on hardware
//
#include <hip/hip_runtime.h>
#include <hip/hip_bf16.h>

typedef __bf16 bf16_t;
typedef __attribute__((ext_vector_type(16))) __bf16 v16bf;
typedef __attribute__((ext_vector_type(8)))  __bf16 v8bf;
typedef __attribute__((ext_vector_type(4)))  __bf16 v4bf;
typedef __attribute__((ext_vector_type(8)))  float  v8f;

#define B_DIM 32
#define T_DIM 2048
#define F_DIM 128
#define TOPK  8

// CDNA5 async global->LDS DMA (ASYNCcnt-tracked), guarded so compile never fails.
#if __has_builtin(__builtin_amdgcn_global_load_async_to_lds_b128) && \
    __has_builtin(__builtin_amdgcn_s_wait_asynccnt)
#define USE_ASYNC_LDS 1
// Prototype (from hipcc diagnostic): (global int4* src, local int4* dst, Ii, Ii)
typedef int v4i_vs __attribute__((vector_size(16)));
typedef __attribute__((address_space(1))) v4i_vs* async_gptr_t;
typedef __attribute__((address_space(3))) v4i_vs* async_lptr_t;
#else
#define USE_ASYNC_LDS 0
#endif

// ---------------------------------------------------------------------------
// Kernel 1: L2-normalize each 128-float row, emit bf16. One wave32 per row.
// ---------------------------------------------------------------------------
__global__ __launch_bounds__(256) void normalize_bf16_kernel(
    const float* __restrict__ feat, bf16_t* __restrict__ fnh, int nRows)
{
  const int row  = blockIdx.x * 8 + (threadIdx.x >> 5);
  const int lane = threadIdx.x & 31;
  if (row >= nRows) return;

  const float4 v = reinterpret_cast<const float4*>(feat + (size_t)row * F_DIM)[lane];
  float ss = v.x * v.x + v.y * v.y + v.z * v.z + v.w * v.w;
  #pragma unroll
  for (int m = 16; m >= 1; m >>= 1) ss += __shfl_xor(ss, m, 32);
  const float inv = 1.0f / sqrtf(ss);

  v4bf o;
  o[0] = (bf16_t)(v.x * inv);
  o[1] = (bf16_t)(v.y * inv);
  o[2] = (bf16_t)(v.z * inv);
  o[3] = (bf16_t)(v.w * inv);
  reinterpret_cast<v4bf*>(fnh + (size_t)row * F_DIM)[lane] = o;
}

// ---------------------------------------------------------------------------
// Kernel 2: fused Gram-GEMM (bf16 WMMA) + softmax row statistics + score.
// 8 waves/block; each wave owns a 16-row M-tile. The s-dimension is streamed
// in 64-row tiles, double-buffered in LDS via async global->LDS DMA.
// Softmax uses the fixed shift 1.0 (valid: cosine sims <= ~1; softmax is
// shift-invariant), so the hot loop is exp+add / exp+fma only:
//   pass 0: Z[r]     = sum_s exp(sim - 1)
//   pass 1: score[r] = mean_s( exp(sim - 1) * (1/Z[r]) )
// ---------------------------------------------------------------------------
__global__ __launch_bounds__(256) void score_kernel(
    const bf16_t* __restrict__ fnh, float* __restrict__ score)
{
  constexpr int LSTR = 136;           // padded LDS row stride (272 B)
  constexpr int TS   = 64;            // s-tile rows per stage
  constexpr int NT   = T_DIM / TS;    // 32 stages per pass
  __shared__ __align__(16) bf16_t tile[2][TS * LSTR];

  const int b       = blockIdx.y;
  const int wave    = threadIdx.x >> 5;
  const int lane    = threadIdx.x & 31;
  const int n16     = lane & 15;
  const int g       = lane >> 4;
  const int rowBase = blockIdx.x * 128 + wave * 16;

  const bf16_t* fb = fnh + (size_t)b * T_DIM * F_DIM;

  // A fragments: rows rowBase..rowBase+15, K=0..127 as four 16x32 bf16 frags.
  // CDNA5 A layout: lane L -> row (L&15); halves 0-7 = K kc+(L>>4)*8..+7,
  //                                       halves 8-15 = K kc+16+(L>>4)*8..+7.
  v16bf a[4];
  {
    const bf16_t* arow = fb + (size_t)(rowBase + n16) * F_DIM;
    #pragma unroll
    for (int kc = 0; kc < 4; ++kc) {
      union { v16bf v; v8bf h[2]; } u;
      u.h[0] = *reinterpret_cast<const v8bf*>(arow + kc * 32 + g * 8);
      u.h[1] = *reinterpret_cast<const v8bf*>(arow + kc * 32 + 16 + g * 8);
      a[kc] = u.v;
    }
  }

  // Staging: 64 rows x 128 bf16 = 1024 x 16B units; 4 units per thread.
  const int srow   = threadIdx.x >> 4;
  const int scol   = (threadIdx.x & 15) * 8;
  auto stage = [&](int buf, int s0) {
    #pragma unroll
    for (int j = 0; j < 4; ++j) {
      const int row = srow + j * 16;
#if USE_ASYNC_LDS
      __builtin_amdgcn_global_load_async_to_lds_b128(
          (async_gptr_t)(fb + (size_t)(s0 + row) * F_DIM + scol),
          (async_lptr_t)&tile[buf][row * LSTR + scol],
          0, 0);
#else
      *reinterpret_cast<v8bf*>(&tile[buf][row * LSTR + scol]) =
          *reinterpret_cast<const v8bf*>(fb + (size_t)(s0 + row) * F_DIM + scol);
      __builtin_prefetch(fb + (size_t)(s0 + TS + row) * F_DIM + scol, 0, 0);
#endif
    }
  };

  float zpart[8], sump[8], invZ[8];
  #pragma unroll
  for (int r = 0; r < 8; ++r) { zpart[r] = 0.0f; sump[r] = 0.0f; invZ[r] = 0.0f; }

  #pragma unroll 1
  for (int pass = 0; pass < 2; ++pass) {
    if (pass == 1) {
      // reduce per-lane partial Z across the 16 N-lanes (layout-matched)
      #pragma unroll
      for (int r = 0; r < 8; ++r) {
        float z = zpart[r];
        z += __shfl_xor(z, 1);
        z += __shfl_xor(z, 2);
        z += __shfl_xor(z, 4);
        z += __shfl_xor(z, 8);
        invZ[r] = 1.0f / z;
      }
    }

    stage(0, 0);  // prime buffer 0

    #pragma unroll 1
    for (int it = 0; it < NT; ++it) {
      const int cur = it & 1;
#if USE_ASYNC_LDS
      __builtin_amdgcn_s_wait_asynccnt(0);
#endif
      __syncthreads();  // tile[cur] filled everywhere; tile[cur^1] fully consumed
      if (it + 1 < NT) stage(cur ^ 1, (it + 1) * TS);

      // 4 N-subtiles of 16 columns; 4 chained WMMAs (K=128) each.
      #pragma unroll
      for (int sn = 0; sn < 4; ++sn) {
        v8f c = {};
        const bf16_t* lrow = &tile[cur][(sn * 16 + n16) * LSTR + g * 8];
        #pragma unroll
        for (int kc = 0; kc < 4; ++kc) {
          union { v16bf v; v8bf h[2]; } u;
          u.h[0] = *reinterpret_cast<const v8bf*>(lrow + kc * 32);
          u.h[1] = *reinterpret_cast<const v8bf*>(lrow + kc * 32 + 16);
          c = __builtin_amdgcn_wmma_f32_16x16x32_bf16(false, a[kc], false, u.v,
                                                      (short)0, c, false, false);
        }
        // C layout: lanes 0-15 -> M=r, lanes 16-31 -> M=8+r, N = lane&15.
        if (pass == 0) {
          #pragma unroll
          for (int r = 0; r < 8; ++r) zpart[r] += __expf(c[r] - 1.0f);
        } else {
          #pragma unroll
          for (int r = 0; r < 8; ++r) sump[r] += __expf(c[r] - 1.0f) * invZ[r];
        }
      }
    }
    __syncthreads();  // drain before the other pass re-primes buffer 0
  }

  // Reduce softmax means across the 16 N-lanes, write per-row score.
  #pragma unroll
  for (int r = 0; r < 8; ++r) {
    float t = sump[r];
    t += __shfl_xor(t, 1);
    t += __shfl_xor(t, 2);
    t += __shfl_xor(t, 4);
    t += __shfl_xor(t, 8);
    if (n16 == r)
      score[(size_t)b * T_DIM + rowBase + g * 8 + r] = t * (1.0f / (float)T_DIM);
  }
}

// ---------------------------------------------------------------------------
// Kernel 3: per-batch top-8 smallest + top-8 largest scores (ties -> lowest
// index, like jax.lax.top_k), then gather feature rows into the output.
// Output layout: hard_k_feat [32,8,128] followed by conf_k_feat [32,8,128].
// ---------------------------------------------------------------------------
__global__ __launch_bounds__(256) void topk_gather_kernel(
    const float* __restrict__ feat, const float* __restrict__ score,
    float* __restrict__ out)
{
  __shared__ float sh[T_DIM];   // -score  (hard: smallest score = largest -score)
  __shared__ float sc[T_DIM];   //  score  (confident: largest score)
  __shared__ float rv[256];
  __shared__ int   ri[256];
  __shared__ int   sel[2 * TOPK];

  const int b   = blockIdx.x;
  const int tid = threadIdx.x;
  const float* sb = score + (size_t)b * T_DIM;
  for (int i = tid; i < T_DIM; i += 256) {
    const float v = sb[i];
    sh[i] = -v;
    sc[i] = v;
  }
  __syncthreads();

  #pragma unroll 1
  for (int slot = 0; slot < 2 * TOPK; ++slot) {
    float* arr = (slot < TOPK) ? sh : sc;
    float bv = -__builtin_inff();
    int   bi = T_DIM;
    for (int i = tid; i < T_DIM; i += 256) {
      const float v = arr[i];       // ascending scan: strict > keeps lowest idx
      if (v > bv) { bv = v; bi = i; }
    }
    rv[tid] = bv; ri[tid] = bi;
    __syncthreads();
    for (int st = 128; st > 0; st >>= 1) {
      if (tid < st) {
        const float ov = rv[tid + st]; const int oi = ri[tid + st];
        if (ov > rv[tid] || (ov == rv[tid] && oi < ri[tid])) { rv[tid] = ov; ri[tid] = oi; }
      }
      __syncthreads();
    }
    if (tid == 0) {
      sel[slot] = ri[0];
      arr[ri[0]] = -__builtin_inff();
    }
    __syncthreads();
  }

  const float* fbat = feat + (size_t)b * T_DIM * F_DIM;
  for (int j = tid; j < 2 * TOPK * F_DIM; j += 256) {
    const int half = j / (TOPK * F_DIM);            // 0 = hard, 1 = confident
    const int rem  = j - half * (TOPK * F_DIM);
    const int k    = rem / F_DIM;
    const int f    = rem - k * F_DIM;
    const int idx  = sel[half * TOPK + k];
    out[(size_t)half * B_DIM * TOPK * F_DIM + (size_t)b * TOPK * F_DIM + rem] =
        fbat[(size_t)idx * F_DIM + f];
  }
}

// ---------------------------------------------------------------------------
extern "C" void kernel_launch(void* const* d_in, const int* in_sizes, int n_in,
                              void* d_out, int out_size, void* d_ws, size_t ws_size,
                              hipStream_t stream)
{
  (void)in_sizes; (void)n_in; (void)out_size; (void)ws_size;

  const float* feat = (const float*)d_in[0];
  bf16_t* fnh  = (bf16_t*)d_ws;                                   // 16 MB bf16 fn
  float*  score = (float*)((char*)d_ws +
                           (size_t)B_DIM * T_DIM * F_DIM * sizeof(bf16_t)); // 256 KB
  float*  out = (float*)d_out;

  const int nRows = B_DIM * T_DIM;
  normalize_bf16_kernel<<<nRows / 8, 256, 0, stream>>>(feat, fnh, nRows);

  dim3 grid(T_DIM / 128, B_DIM);
  score_kernel<<<grid, 256, 0, stream>>>(fnh, score);

  topk_gather_kernel<<<B_DIM, 256, 0, stream>>>(feat, score, out);
}
